// MixtureOfRecursions_12326556139642
// MI455X (gfx1250) — compile-verified, matrix operand order checked by reading, not run on hardware
//
#include <hip/hip_runtime.h>
#include <hip/hip_bf16.h>
#include <math.h>

// ---------------- model dims ----------------
#define VOC   50257
#define EMB   768
#define NH    12
#define HD    64
#define FF    3072
#define NL    4
#define RHID  128
#define NDEP  4
#define NB    2
#define SEQ   1024
#define TOK   (NB * SEQ)   // 2048

typedef __attribute__((ext_vector_type(16))) __bf16 v16bf;
typedef __attribute__((ext_vector_type(8)))  __bf16 v8bf;
typedef __attribute__((ext_vector_type(8)))  float  v8f;
typedef __attribute__((ext_vector_type(4)))  unsigned int v4u;
typedef __attribute__((ext_vector_type(8)))  int v8i;
typedef __attribute__((ext_vector_type(4)))  int v4i;

__device__ __forceinline__ __bf16 f2bf(float x) {
  unsigned u = __builtin_bit_cast(unsigned, x);
  unsigned r = (u + 0x7FFFu + ((u >> 16) & 1u)) >> 16;
  unsigned short s = (unsigned short)r;
  return __builtin_bit_cast(__bf16, s);
}
__device__ __forceinline__ float bf2f(__bf16 h) {
  unsigned short s = __builtin_bit_cast(unsigned short, h);
  unsigned u = ((unsigned)s) << 16;
  return __builtin_bit_cast(float, u);
}
__device__ __forceinline__ void store_out(float* p, float v)  { *p = v; }
__device__ __forceinline__ void store_out(__bf16* p, float v) { *p = f2bf(v); }

// ---------------- Tensor Data Mover: 2-D bf16 tile -> padded LDS ----------------
// Loads a (tile_rows x 32) bf16 tile whose rows are `row_stride` elements apart in
// global memory into LDS with 16B of padding after every 64B row (stride 40 bf16).
// Rows/cols beyond (rem_rows, rem_k) are zero-filled by TDM OOB handling.
__device__ __forceinline__ void tdm_load_tile(unsigned lds_addr, const __bf16* gptr,
                                              unsigned rem_k, unsigned rem_rows,
                                              unsigned tile_rows, unsigned row_stride) {
  unsigned long long ga = (unsigned long long)(uintptr_t)gptr;
  v4u g0;
  g0[0] = 1u;                                        // count=1, user descriptor
  g0[1] = lds_addr;                                  // LDS byte address
  g0[2] = (unsigned)(ga & 0xFFFFFFFFu);              // global_addr[31:0]
  g0[3] = (unsigned)((ga >> 32) & 0x01FFFFFFu)       // global_addr[56:32]
        | (2u << 30);                                // type = 2 (image/tensor)
  v8i g1;
  unsigned w0 = (1u << 16)                           // data_size = 1 -> 2 bytes
              | (1u << 20)                           // pad_enable
              | (3u << 22)                           // pad_interval: 16 DWORDs (64B = one row)
              | (3u << 25);                          // pad_amount: 4 DWORDs (16B)
  g1[0] = (int)w0;
  g1[1] = (int)((rem_k & 0xFFFFu) << 16);            // [47:32]=atomic_addr=0, [63:48]=dim0 lo
  g1[2] = (int)(((rem_k >> 16) & 0xFFFFu) | ((rem_rows & 0xFFFFu) << 16)); // dim0 hi | dim1 lo
  g1[3] = (int)(((rem_rows >> 16) & 0xFFFFu) | (32u << 16));               // dim1 hi | tile_dim0=32
  g1[4] = (int)(tile_rows & 0xFFFFu);                // tile_dim1 | tile_dim2=0
  g1[5] = (int)row_stride;                           // tensor_dim0_stride lo32
  g1[6] = 0;                                         // stride hi | dim1_stride lo
  g1[7] = 0;
  v4i g2 = {0, 0, 0, 0};
  v4i g3 = {0, 0, 0, 0};
#if __clang_major__ >= 23
  v8i g4 = {0, 0, 0, 0, 0, 0, 0, 0};
  __builtin_amdgcn_tensor_load_to_lds(g0, g1, g2, g3, g4, 0);
#else
  __builtin_amdgcn_tensor_load_to_lds(g0, g1, g2, g3, 0);
#endif
}

// ---------------- f32 -> bf16 conversion ----------------
__global__ __launch_bounds__(256) void f2bf_kernel(const float* __restrict__ x,
                                                   __bf16* __restrict__ y, size_t n) {
  size_t i = (size_t)blockIdx.x * 256 + threadIdx.x;
  if (i < n) y[i] = f2bf(x[i]);
}

// ---------------- embedding ----------------
__global__ __launch_bounds__(256) void embed_kernel(const int* __restrict__ ids,
                                                    const float* __restrict__ tok,
                                                    const float* __restrict__ pos,
                                                    float* __restrict__ h,
                                                    __bf16* __restrict__ hbf) {
  size_t i = (size_t)blockIdx.x * 256 + threadIdx.x;
  if (i >= (size_t)TOK * EMB) return;
  int e = (int)(i % EMB);
  int t = (int)(i / EMB);
  int s = t % SEQ;
  float v = tok[(size_t)ids[t] * EMB + e] + pos[(size_t)s * EMB + e];
  h[i] = v;
  hbf[i] = f2bf(v);
}

// ---------------- layernorm (one block per row of 768) ----------------
__global__ __launch_bounds__(256) void ln_kernel(const float* __restrict__ x,
                                                 const float* __restrict__ g,
                                                 const float* __restrict__ b,
                                                 __bf16* __restrict__ y) {
  int row = blockIdx.x;
  int tid = threadIdx.x;
  const float* xr = x + (size_t)row * EMB;
  float vals[3];
  float s = 0.f;
  for (int i = 0; i < 3; ++i) { vals[i] = xr[tid + i * 256]; s += vals[i]; }
  __shared__ float red[256];
  red[tid] = s;
  __syncthreads();
  for (int off = 128; off > 0; off >>= 1) {
    if (tid < off) red[tid] += red[tid + off];
    __syncthreads();
  }
  float mu = red[0] * (1.0f / EMB);
  __syncthreads();
  float sq = 0.f;
  for (int i = 0; i < 3; ++i) { float d = vals[i] - mu; sq += d * d; }
  red[tid] = sq;
  __syncthreads();
  for (int off = 128; off > 0; off >>= 1) {
    if (tid < off) red[tid] += red[tid + off];
    __syncthreads();
  }
  float rstd = rsqrtf(red[0] * (1.0f / EMB) + 1e-5f);
  for (int i = 0; i < 3; ++i) {
    int c = tid + i * 256;
    y[(size_t)row * EMB + c] = f2bf((vals[i] - mu) * rstd * g[c] + b[c]);
  }
}

// ---------------- bf16 WMMA GEMM: C[M,N] = act(A[M,K] @ B[N,K]^T + bias) + resid ----------------
// TDM double-buffered tiles; block tile 128x64; each wave computes 32x32 (4 WMMAs/step).
// Requires M % 128 == 0 and K % 32 == 0 (true for every call site); N may be ragged.
// ACT: 0 = none, 1 = relu, 2 = exact gelu
#define BM 128
#define BN 64
#define BK 32
#define LDSK 40   // padded row stride (bf16 units): 32 data + 8 pad, keeps 16B alignment

template <typename OutT, int ACT>
__global__ __launch_bounds__(256) void gemm_bf16_wmma(
    const __bf16* __restrict__ A, const __bf16* __restrict__ Bw,
    const float* __restrict__ bias, const float* __restrict__ resid,
    OutT* __restrict__ C, int M, int N, int K) {
  __shared__ __align__(16) __bf16 As[2][BM * LDSK];
  __shared__ __align__(16) __bf16 Bs[2][BN * LDSK];

  const int tid  = threadIdx.x;
  const int lane = tid & 31;
  const int wave = tid >> 5;
  const int m0 = blockIdx.y * BM;
  const int n0 = blockIdx.x * BN;
  const int rw = (wave & 3) << 5;   // wave rows: 0,32,64,96 (two 16-tiles)
  const int cw = (wave >> 2) << 5;  // wave cols: 0 or 32     (two 16-tiles)

  v8f acc00 = {}, acc01 = {}, acc10 = {}, acc11 = {};

  const unsigned remN = (unsigned)(N - n0);
  const int nk = K / BK;

  // prologue: stage 0 via Tensor Data Mover (wave 0 issues; EXEC ignored by TDM)
  if (wave == 0) {
    tdm_load_tile((unsigned)(uintptr_t)&As[0][0], A + (size_t)m0 * K,
                  (unsigned)K, (unsigned)(M - m0), BM, (unsigned)K);
    tdm_load_tile((unsigned)(uintptr_t)&Bs[0][0], Bw + (size_t)n0 * K,
                  (unsigned)K, remN, BN, (unsigned)K);
  }

  const int half = (lane < 16) ? 0 : 8;   // ISA 16-bit A-layout K-half select
  const int l15  = lane & 15;

  for (int i = 0; i < nk; ++i) {
    if (wave == 0) {
      if (i + 1 < nk) {
        int k1 = (i + 1) * BK;
        int buf = (i + 1) & 1;
        tdm_load_tile((unsigned)(uintptr_t)&As[buf][0], A + (size_t)m0 * K + k1,
                      (unsigned)(K - k1), (unsigned)(M - m0), BM, (unsigned)K);
        tdm_load_tile((unsigned)(uintptr_t)&Bs[buf][0], Bw + (size_t)n0 * K + k1,
                      (unsigned)(K - k1), remN, BN, (unsigned)K);
        __builtin_amdgcn_s_wait_tensorcnt(2);  // stage i complete (TDM in-order)
      } else {
        __builtin_amdgcn_s_wait_tensorcnt(0);
      }
    }
    __syncthreads();

    const __bf16* Ab = &As[i & 1][0];
    const __bf16* Bb = &Bs[i & 1][0];
    auto ldfrag = [&](const __bf16* base, int row) -> v16bf {
      v8bf lo = *(const v8bf*)(base + row * LDSK + half);
      v8bf hi = *(const v8bf*)(base + row * LDSK + half + 16);
      return __builtin_shufflevector(lo, hi, 0, 1, 2, 3, 4, 5, 6, 7,
                                     8, 9, 10, 11, 12, 13, 14, 15);
    };
    v16bf a0 = ldfrag(Ab, rw + l15);
    v16bf a1 = ldfrag(Ab, rw + 16 + l15);
    v16bf b0 = ldfrag(Bb, cw + l15);
    v16bf b1 = ldfrag(Bb, cw + 16 + l15);

    acc00 = __builtin_amdgcn_wmma_f32_16x16x32_bf16(false, a0, false, b0, (short)0, acc00, false, false);
    acc01 = __builtin_amdgcn_wmma_f32_16x16x32_bf16(false, a0, false, b1, (short)0, acc01, false, false);
    acc10 = __builtin_amdgcn_wmma_f32_16x16x32_bf16(false, a1, false, b0, (short)0, acc10, false, false);
    acc11 = __builtin_amdgcn_wmma_f32_16x16x32_bf16(false, a1, false, b1, (short)0, acc11, false, false);
    __syncthreads();  // all reads done before buffer is re-filled
  }

  // epilogue: C layout -> VGPR r: M = r (lanes 0-15) / r+8 (lanes 16-31), N = lane%16
  const int rb = m0 + rw + ((lane & 16) ? 8 : 0);
  const int c0 = n0 + cw + l15;
  const int c1 = c0 + 16;
  float bias0 = (bias && c0 < N) ? bias[c0] : 0.f;
  float bias1 = (bias && c1 < N) ? bias[c1] : 0.f;
  auto emit = [&](int row, int col, float v, float bv) {
    if (col >= N) return;
    v += bv;
    if (ACT == 1) v = v > 0.f ? v : 0.f;
    if (ACT == 2) v = 0.5f * v * (1.0f + erff(v * 0.70710678118654752f));
    if (resid) v += resid[(size_t)row * N + col];
    store_out(C + (size_t)row * N + col, v);
  };
  for (int r = 0; r < 8; ++r) {
    emit(rb + r,      c0, acc00[r], bias0);
    emit(rb + r,      c1, acc01[r], bias1);
    emit(rb + 16 + r, c0, acc10[r], bias0);
    emit(rb + 16 + r, c1, acc11[r], bias1);
  }
}

// ---------------- attention: one wave32 per (b, h, q); online softmax over all S keys ----------------
// NOTE: reference ADDS a 0/1 mask to scores (not -inf masking), so softmax spans all keys.
__global__ __launch_bounds__(256) void attn_kernel(const float* __restrict__ qkv,
                                                   const int* __restrict__ depths,
                                                   int depth,
                                                   __bf16* __restrict__ ctx) {
  int wid  = blockIdx.x * 8 + (threadIdx.x >> 5);
  int lane = threadIdx.x & 31;
  int q = wid % SEQ;
  int h = (wid / SEQ) % NH;
  int b = wid / (SEQ * NH);

  const size_t row3E = (size_t)3 * EMB;
  const float* qp = qkv + ((size_t)(b * SEQ + q)) * row3E + h * HD;
  float q0 = qp[lane * 2];
  float q1 = qp[lane * 2 + 1];
  bool dq = depths[b * SEQ + q] >= depth;

  float m = -1e30f, l = 0.f, a0 = 0.f, a1 = 0.f;
  for (int kk = 0; kk < SEQ; ++kk) {
    const float* kp = qkv + ((size_t)(b * SEQ + kk)) * row3E + EMB + h * HD;
    const float* vp = kp + EMB;
    __builtin_prefetch(vp + lane * 2, 0, 1);
    float s = q0 * kp[lane * 2] + q1 * kp[lane * 2 + 1];
    for (int off = 16; off > 0; off >>= 1) s += __shfl_xor(s, off, 32);
    s *= 0.125f;  // 1/sqrt(64)
    if (kk <= q && dq && depths[b * SEQ + kk] >= depth) s += 1.0f;
    float mn = fmaxf(m, s);
    float sc = __expf(m - mn);
    float p  = __expf(s - mn);
    l = l * sc + p;
    a0 = a0 * sc + p * vp[lane * 2];
    a1 = a1 * sc + p * vp[lane * 2 + 1];
    m = mn;
  }
  float inv = 1.0f / l;
  size_t o = ((size_t)(b * SEQ + q)) * EMB + h * HD + lane * 2;
  ctx[o]     = f2bf(a0 * inv);
  ctx[o + 1] = f2bf(a1 * inv);
}

// ---------------- router fc2 (N=4): logits, argmax depth, softmax accumulation ----------------
__global__ __launch_bounds__(256) void router2_kernel(const __bf16* __restrict__ r1,
                                                      const float* __restrict__ w2,
                                                      const float* __restrict__ b2,
                                                      float* __restrict__ out_logits,
                                                      float* __restrict__ out_depths_f,
                                                      int* __restrict__ depths_i,
                                                      float* __restrict__ acc) {
  int t = blockIdx.x * 256 + threadIdx.x;
  if (t >= TOK) return;
  float lg[NDEP];
  for (int d = 0; d < NDEP; ++d) lg[d] = b2[d];
  for (int j = 0; j < RHID; ++j) {
    float x = bf2f(r1[(size_t)t * RHID + j]);
    for (int d = 0; d < NDEP; ++d) lg[d] += x * w2[d * RHID + j];
  }
  int arg = 0;
  float mx = lg[0];
  for (int d = 1; d < NDEP; ++d) if (lg[d] > mx) { mx = lg[d]; arg = d; }  // first-max ties
  int depth = arg + 1;
  for (int d = 0; d < NDEP; ++d) out_logits[(size_t)t * NDEP + d] = lg[d];
  out_depths_f[t] = (float)depth;
  depths_i[t] = depth;
  float s = 0.f, p[NDEP];
  for (int d = 0; d < NDEP; ++d) { p[d] = __expf(lg[d] - mx); s += p[d]; }
  float invs = 1.0f / s;
  for (int d = 0; d < NDEP; ++d) atomicAdd(&acc[d], p[d] * invs);
  atomicAdd(&acc[NDEP], (float)depth);
}

__global__ void router_loss_kernel(const float* __restrict__ acc, float* __restrict__ out_loss) {
  if (blockIdx.x == 0 && threadIdx.x == 0) {
    const float u = 1.0f / NDEP;
    float lb = 0.f;
    for (int d = 0; d < NDEP; ++d) {
      float mp = acc[d] / (float)TOK;
      lb += u * (__logf(u) - __logf(mp));
    }
    lb *= (1.0f / NDEP);
    float sparsity = (acc[NDEP] / (float)TOK) / (float)NDEP;
    out_loss[0] = 0.01f * lb + 0.001f * sparsity;
  }
}

// ---------------- host side ----------------
static inline unsigned cdiv(size_t a, size_t b) { return (unsigned)((a + b - 1) / b); }

extern "C" void kernel_launch(void* const* d_in, const int* in_sizes, int n_in,
                              void* d_out, int out_size, void* d_ws, size_t ws_size,
                              hipStream_t stream) {
  (void)in_sizes; (void)n_in; (void)out_size; (void)ws_size;
  // inputs in setup_inputs() order
  const int*   ids     = (const int*)  d_in[0];
  const float* tok_emb = (const float*)d_in[1];
  const float* pos_emb = (const float*)d_in[2];
  const float* rw1     = (const float*)d_in[3];
  const float* rb1     = (const float*)d_in[4];
  const float* rw2     = (const float*)d_in[5];
  const float* rb2     = (const float*)d_in[6];
  const float* wqkv    = (const float*)d_in[7];
  const float* bqkv    = (const float*)d_in[8];
  const float* wo      = (const float*)d_in[9];
  const float* bo      = (const float*)d_in[10];
  const float* ln1g    = (const float*)d_in[11];
  const float* ln1b    = (const float*)d_in[12];
  const float* ln2g    = (const float*)d_in[13];
  const float* ln2b    = (const float*)d_in[14];
  const float* w1      = (const float*)d_in[15];
  const float* b1      = (const float*)d_in[16];
  const float* w2      = (const float*)d_in[17];
  const float* b2      = (const float*)d_in[18];
  const float* lnfg    = (const float*)d_in[19];
  const float* lnfb    = (const float*)d_in[20];
  const float* lm      = (const float*)d_in[21];

  // output layout: logits [TOK,VOC] | depths [TOK] | loss [1] | router_logits [TOK,NDEP]
  float* out_logits   = (float*)d_out;
  float* out_depths_f = out_logits + (size_t)TOK * VOC;
  float* out_loss     = out_depths_f + TOK;
  float* out_rlogits  = out_loss + 1;

  // workspace carve-up (256B aligned)
  char* base = (char*)d_ws;
  size_t off = 0;
  auto take = [&](size_t bytes) -> char* {
    char* p = base + off;
    off = (off + bytes + 255) & ~(size_t)255;
    return p;
  };
  const size_t n_wqkv = (size_t)NL * 3 * EMB * EMB;
  const size_t n_wo   = (size_t)NL * EMB * EMB;
  const size_t n_w1   = (size_t)NL * FF * EMB;
  const size_t n_w2   = (size_t)NL * EMB * FF;
  const size_t n_lm   = (size_t)VOC * EMB;
  const size_t n_rw1  = (size_t)RHID * EMB;

  __bf16* wqkv_bf = (__bf16*)take(n_wqkv * 2);
  __bf16* wo_bf   = (__bf16*)take(n_wo * 2);
  __bf16* w1_bf   = (__bf16*)take(n_w1 * 2);
  __bf16* w2_bf   = (__bf16*)take(n_w2 * 2);
  __bf16* lm_bf   = (__bf16*)take(n_lm * 2);
  __bf16* rw1_bf  = (__bf16*)take(n_rw1 * 2);

  float*  h       = (float*) take((size_t)TOK * EMB * 4);
  __bf16* h_bf    = (__bf16*)take((size_t)TOK * EMB * 2);
  __bf16* hn_bf   = (__bf16*)take((size_t)TOK * EMB * 2);
  float*  qkv     = (float*) take((size_t)TOK * 3 * EMB * 4);
  __bf16* ctx_bf  = (__bf16*)take((size_t)TOK * EMB * 2);
  __bf16* ff1_bf  = (__bf16*)take((size_t)TOK * FF * 2);
  __bf16* r1_bf   = (__bf16*)take((size_t)TOK * RHID * 2);
  int*    depths  = (int*)   take((size_t)TOK * 4);
  float*  acc     = (float*) take(8 * 4);

  hipMemsetAsync(acc, 0, 8 * sizeof(float), stream);

  // weight conversion f32 -> bf16 (per-launch, deterministic)
  auto cvt = [&](const float* s, __bf16* d, size_t n) {
    f2bf_kernel<<<cdiv(n, 256), 256, 0, stream>>>(s, d, n);
  };
  cvt(wqkv, wqkv_bf, n_wqkv);
  cvt(wo, wo_bf, n_wo);
  cvt(w1, w1_bf, n_w1);
  cvt(w2, w2_bf, n_w2);
  cvt(lm, lm_bf, n_lm);
  cvt(rw1, rw1_bf, n_rw1);

  // embedding
  embed_kernel<<<cdiv((size_t)TOK * EMB, 256), 256, 0, stream>>>(ids, tok_emb, pos_emb, h, h_bf);

  // router: fc1 (relu, bf16 out) -> fc2 -> loss
  {
    dim3 g(cdiv(RHID, BN), cdiv(TOK, BM));
    gemm_bf16_wmma<__bf16, 1><<<g, 256, 0, stream>>>(h_bf, rw1_bf, rb1, nullptr, r1_bf, TOK, RHID, EMB);
  }
  router2_kernel<<<cdiv(TOK, 256), 256, 0, stream>>>(r1_bf, rw2, rb2, out_rlogits, out_depths_f, depths, acc);
  router_loss_kernel<<<1, 1, 0, stream>>>(acc, out_loss);

  // transformer layers (depth 1..4, layer index = depth-1)
  for (int depth = 1; depth <= NDEP; ++depth) {
    int li = (depth - 1) % NL;
    const __bf16* Wqkv = wqkv_bf + (size_t)li * 3 * EMB * EMB;
    const __bf16* Wo   = wo_bf + (size_t)li * EMB * EMB;
    const __bf16* W1   = w1_bf + (size_t)li * FF * EMB;
    const __bf16* W2   = w2_bf + (size_t)li * EMB * FF;

    ln_kernel<<<TOK, 256, 0, stream>>>(h, ln1g + li * EMB, ln1b + li * EMB, hn_bf);
    {
      dim3 g(cdiv(3 * EMB, BN), cdiv(TOK, BM));
      gemm_bf16_wmma<float, 0><<<g, 256, 0, stream>>>(hn_bf, Wqkv, bqkv + li * 3 * EMB, nullptr, qkv, TOK, 3 * EMB, EMB);
    }
    attn_kernel<<<(NB * NH * SEQ) / 8, 256, 0, stream>>>(qkv, depths, depth, ctx_bf);
    {
      dim3 g(cdiv(EMB, BN), cdiv(TOK, BM));
      gemm_bf16_wmma<float, 0><<<g, 256, 0, stream>>>(ctx_bf, Wo, bo + li * EMB, h, h, TOK, EMB, EMB);
    }
    ln_kernel<<<TOK, 256, 0, stream>>>(h, ln2g + li * EMB, ln2b + li * EMB, hn_bf);
    {
      dim3 g(cdiv(FF, BN), cdiv(TOK, BM));
      gemm_bf16_wmma<__bf16, 2><<<g, 256, 0, stream>>>(hn_bf, W1, b1 + li * FF, nullptr, ff1_bf, TOK, FF, EMB);
    }
    {
      dim3 g(cdiv(EMB, BN), cdiv(TOK, BM));
      gemm_bf16_wmma<float, 0><<<g, 256, 0, stream>>>(ff1_bf, W2, b2 + li * EMB, h, h, TOK, EMB, FF);
    }
  }

  // final LN + LM head
  ln_kernel<<<TOK, 256, 0, stream>>>(h, lnfg, lnfb, hn_bf);
  {
    dim3 g(cdiv(VOC, BN), cdiv(TOK, BM));
    gemm_bf16_wmma<float, 0><<<g, 256, 0, stream>>>(hn_bf, lm_bf, nullptr, nullptr, out_logits, TOK, VOC, EMB);
  }
}
